// LSTM_51866025067221
// MI455X (gfx1250) — matrix operand order repaired and checked by reading, not compiled, not run on hardware
//
#include <hip/hip_runtime.h>
#include <hip/hip_bf16.h>

// LSTM: T=2048, B=32, I=H=512, 4H=2048, L=2
// Phase 1 per chunk: gi = X @ W_ih^T + bias   (WMMA f16 GEMM, f32 accum)
// Phase 2 per chunk: persistent cooperative scan, 32 WGs (one per 16-wide h
//   slice), 8 waves/WG = {4 gates} x {2 M-tiles}, one v_wmma_f32_16x16x32_f16
//   tile per wave per K-step, grid barrier between timesteps.
// Data movement: global->LDS staging uses CDNA5 async-to-LDS ops
//   (GLOBAL_LOAD_ASYNC_TO_LDS_B128, ASYNCcnt + s_wait_asynccnt) when exposed.

#define LSTM_T   2048
#define LSTM_B   32
#define LSTM_K   512      // I == H
#define LSTM_G   2048     // 4H
#define LSTM_L   2

typedef __attribute__((ext_vector_type(16))) _Float16 v16h;
typedef __attribute__((ext_vector_type(8)))  float    v8f;

#if __has_builtin(__builtin_amdgcn_global_load_async_to_lds_b128)
#define HAVE_ASYNC_LDS 1
typedef __attribute__((__vector_size__(16))) int v4i_t;
typedef __attribute__((address_space(1))) v4i_t* gv4i_p;   // global int4*
typedef __attribute__((address_space(3))) v4i_t* lv4i_p;   // LDS int4*
#else
#define HAVE_ASYNC_LDS 0
#endif

// copy 16 bytes global -> LDS (async path writes LDS directly, no VGPR round trip)
__device__ __forceinline__ void stage16(const void* gsrc, void* ldst) {
#if HAVE_ASYNC_LDS
    __builtin_amdgcn_global_load_async_to_lds_b128((gv4i_p)gsrc, (lv4i_p)ldst, 0, 0);
#else
    *(uint4*)ldst = *(const uint4*)gsrc;
#endif
}

__device__ __forceinline__ void stage_wait() {
#if HAVE_ASYNC_LDS
#if __has_builtin(__builtin_amdgcn_s_wait_asynccnt)
    __builtin_amdgcn_s_wait_asynccnt(0);
#else
    asm volatile("s_wait_asynccnt 0x0" ::: "memory");
#endif
#endif
}

__device__ __forceinline__ float sigm(float v) {
    return 1.0f / (1.0f + __expf(-v));
}

__device__ __forceinline__ void grid_barrier(unsigned* ctr, unsigned* gen, int nwg) {
    __syncthreads();
    __threadfence();                       // release: make h_state stores visible
    if (threadIdx.x == 0) {
        __hip_atomic_fetch_add(ctr, 1u, __ATOMIC_RELAXED, __HIP_MEMORY_SCOPE_AGENT);
        unsigned target = (*gen + 1u) * (unsigned)nwg;
        while (__hip_atomic_load(ctr, __ATOMIC_RELAXED, __HIP_MEMORY_SCOPE_AGENT) < target) {
            __builtin_amdgcn_s_sleep(2);
        }
    }
    __syncthreads();
    __threadfence();                       // acquire: invalidate WGP$ before reading peers' h
    ++*gen;
}

// ---------------------------------------------------------------- weights f32->f16
__global__ __launch_bounds__(256) void cvt_f16_kernel(const float* __restrict__ src,
                                                      _Float16* __restrict__ dst, int n) {
    int i = blockIdx.x * blockDim.x + threadIdx.x;
    if (i < n) dst[i] = (_Float16)src[i];
}

// ---------------------------------------------------------------- gi GEMM
// X: [rows,512] f32 ; W: [2048,512] f16 row-major (B tile in [n][k] form)
// gi: [rows,2048] f32 = X @ W^T + bias
// grid: (rows/64, 16), block 256 (8 waves); wave tile 32x32 = 2x2 wmma tiles
__global__ __launch_bounds__(256) void gemm_gi_kernel(const float* __restrict__ X,
                                                      const _Float16* __restrict__ W,
                                                      const float* __restrict__ bias,
                                                      float* __restrict__ gi,
                                                      unsigned* barrier_ctr) {
    __shared__ __align__(16) _Float16 As[64 * 32];   // [row][k]
    __shared__ __align__(16) _Float16 Bs[128 * 32];  // [n][k]

    if (blockIdx.x == 0 && blockIdx.y == 0 && threadIdx.x == 0)
        *barrier_ctr = 0u;                 // arm the scan kernel's grid barrier

    const int tid  = threadIdx.x;
    const int lane = tid & 31;
    const int wave = tid >> 5;
    const int wm   = wave & 1;             // 2 waves along M
    const int wn   = wave >> 1;            // 4 waves along N
    const int row0 = blockIdx.x * 64;
    const int col0 = blockIdx.y * 128;
    const int half = lane >> 4;            // lane group (0: lanes0-15, 1: lanes16-31)
    const int ml   = lane & 15;

    v8f acc[2][2] = {};

    for (int k0 = 0; k0 < LSTM_K; k0 += 32) {
        // stage A (f32 -> f16, packed): thread -> (row, 8-wide k group)
        {
            int ar = tid >> 2;                 // 0..63
            int kg = (tid & 3) * 8;            // 0,8,16,24
            const float* xp = X + (size_t)(row0 + ar) * LSTM_K + k0 + kg;
            float4 f0 = ((const float4*)xp)[0];
            float4 f1 = ((const float4*)xp)[1];
            union { _Float16 h[8]; uint4 u; } pk;
            pk.h[0] = (_Float16)f0.x; pk.h[1] = (_Float16)f0.y;
            pk.h[2] = (_Float16)f0.z; pk.h[3] = (_Float16)f0.w;
            pk.h[4] = (_Float16)f1.x; pk.h[5] = (_Float16)f1.y;
            pk.h[6] = (_Float16)f1.z; pk.h[7] = (_Float16)f1.w;
            *(uint4*)&As[ar * 32 + kg] = pk.u;
        }
        // stage B (already f16): 128x32 = 4096 halfs, 16 contiguous per thread
        {
            int idx = tid * 16;
            int br = idx >> 5, bk = idx & 31;
            const _Float16* wp = W + (size_t)(col0 + br) * LSTM_K + k0 + bk;
            stage16(wp,     &Bs[idx]);
            stage16(wp + 8, &Bs[idx + 8]);
        }
        stage_wait();
        __syncthreads();

        v16h afrag[2], bfrag[2];
        #pragma unroll
        for (int i = 0; i < 2; ++i) {
            const _Float16* ap = &As[(wm * 32 + i * 16 + ml) * 32];
            #pragma unroll
            for (int e = 0; e < 16; e += 2) {   // A 16x32: k = 2v+16*half
                int v = e >> 1;
                int k = 2 * v + 16 * half;
                afrag[i][e]     = ap[k];
                afrag[i][e + 1] = ap[k + 1];
            }
            const _Float16* bp = &Bs[(wn * 32 + i * 16 + ml) * 32 + 16 * half];
            #pragma unroll
            for (int e = 0; e < 16; ++e) bfrag[i][e] = bp[e];  // B 32x16: k = e + 16*half
        }
        #pragma unroll
        for (int i = 0; i < 2; ++i)
            #pragma unroll
            for (int j = 0; j < 2; ++j)
                acc[i][j] = __builtin_amdgcn_wmma_f32_16x16x32_f16(
                    false, afrag[i], false, bfrag[j], (short)0, acc[i][j], false, false);
        __syncthreads();
    }

    // epilogue: + bias, store f32.  D layout: m = e + 8*half, n = lane%16
    #pragma unroll
    for (int i = 0; i < 2; ++i)
        #pragma unroll
        for (int j = 0; j < 2; ++j) {
            int col = col0 + wn * 32 + j * 16 + ml;
            float bv = bias[col];
            #pragma unroll
            for (int e = 0; e < 8; ++e) {
                int row = row0 + wm * 32 + i * 16 + e + 8 * half;
                gi[(size_t)row * LSTM_G + col] = acc[i][j][e] + bv;
            }
        }
}

// ---------------------------------------------------------------- recurrent scan
// grid: 32 WGs (h slice of 16 cols each), block 256 = 8 waves
// wave -> gate q = wave>>1 (i,f,g,o), M-tile mt = wave&1 (batch rows 16mt..)
__global__ __launch_bounds__(256) void lstm_scan_kernel(const float* __restrict__ gi,  // [CH,32,2048]
                                                        const _Float16* __restrict__ Whh, // [2048,512]
                                                        _Float16* h_state,   // [32,512] f16
                                                        float* c_state,      // [32,512] f32
                                                        float* __restrict__ y, // chunk base [CH,32,512]
                                                        unsigned* barrier_ctr,
                                                        int nsteps, int init) {
    __shared__ __align__(16) _Float16 hs[LSTM_B * LSTM_K];   // 32 KB staged h
    __shared__ __align__(16) _Float16 ws_whh[64 * LSTM_K];   // 64 KB W_hh slice (rows q*16+j)
    __shared__ float gates[4 * LSTM_B * 16];                 // 8 KB
    __shared__ float cbuf[LSTM_B * 16];                      // 2 KB, persists across steps

    const int tid   = threadIdx.x;
    const int lane  = tid & 31;
    const int wave  = tid >> 5;
    const int q     = wave >> 1;
    const int mt    = wave & 1;
    const int hcol0 = blockIdx.x * 16;
    const int n0    = q * LSTM_K + hcol0;       // gate column base in [0,2048)
    const int half  = lane >> 4;
    const int ml    = lane & 15;
    unsigned gen = 0;

    // stage this WG's W_hh slice once: local row lr = q*16+j <-> global row q*512+hcol0+j
    // 64 rows x 512 halfs = 32768 halfs; 128 per thread = 16 x 16B
    #pragma unroll
    for (int r = 0; r < 16; ++r) {
        int idx = (tid + r * 256) * 8;           // half index, 16B granules
        int lr = idx >> 9;                       // /512
        int lk = idx & 511;
        int grow = (lr >> 4) * LSTM_K + hcol0 + (lr & 15);
        stage16(Whh + (size_t)grow * LSTM_K + lk, &ws_whh[lr * LSTM_K + lk]);
    }

    for (int idx = tid; idx < LSTM_B * 16; idx += 256) {
        int b2 = idx >> 4, hh = idx & 15;
        cbuf[idx] = init ? 0.0f : c_state[b2 * LSTM_K + hcol0 + hh];
        if (init) h_state[b2 * LSTM_K + hcol0 + hh] = (_Float16)0.0f;
    }
    stage_wait();
    grid_barrier(barrier_ctr, &gen, 32);

    for (int t = 0; t < nsteps; ++t) {
        // stage full h (32x512 f16 = 2048 x 16B) into LDS
        #pragma unroll
        for (int r = 0; r < 8; ++r) {
            int idx = tid + r * 256;
            stage16((const uint4*)h_state + idx, (uint4*)hs + idx);
        }

        const float* gp = gi + (size_t)t * LSTM_B * LSTM_G;
        if (t + 1 < nsteps)   // hint next step's gi tile toward the caches
            __builtin_prefetch(gp + (size_t)LSTM_B * LSTM_G + (size_t)(mt * 16) * LSTM_G + n0, 0, 1);

        v8f acc;
        #pragma unroll
        for (int e = 0; e < 8; ++e)
            acc[e] = gp[(size_t)(mt * 16 + e + 8 * half) * LSTM_G + n0 + ml];

        stage_wait();
        __syncthreads();

        for (int k0 = 0; k0 < LSTM_K; k0 += 32) {
            v16h afrag, bfrag;
            const _Float16* ap = &hs[(mt * 16 + ml) * LSTM_K + k0];
            #pragma unroll
            for (int e = 0; e < 16; e += 2) {
                int v = e >> 1;
                int k = 2 * v + 16 * half;
                afrag[e]     = ap[k];
                afrag[e + 1] = ap[k + 1];
            }
            const _Float16* bp = &ws_whh[(q * 16 + ml) * LSTM_K + k0 + 16 * half];
            #pragma unroll
            for (int e = 0; e < 16; ++e) bfrag[e] = bp[e];
            acc = __builtin_amdgcn_wmma_f32_16x16x32_f16(
                false, afrag, false, bfrag, (short)0, acc, false, false);
        }

        // gates tile -> LDS (m = e + 8*half, n = ml)
        float* gq = &gates[(q * LSTM_B + mt * 16) * 16];
        #pragma unroll
        for (int e = 0; e < 8; ++e) gq[(e + 8 * half) * 16 + ml] = acc[e];
        __syncthreads();

        // elementwise cell update: 512 elems, 2 per thread
        #pragma unroll
        for (int r = 0; r < 2; ++r) {
            int idx = tid + r * 256;
            int b2 = idx >> 4, hh = idx & 15;
            float ig = gates[(0 * LSTM_B + b2) * 16 + hh];
            float fg = gates[(1 * LSTM_B + b2) * 16 + hh];
            float gg = gates[(2 * LSTM_B + b2) * 16 + hh];
            float og = gates[(3 * LSTM_B + b2) * 16 + hh];
            float c_old = cbuf[idx];
            float c_new = sigm(fg) * c_old + sigm(ig) * tanhf(gg);
            float h_new = sigm(og) * tanhf(c_new);
            cbuf[idx] = c_new;
            h_state[b2 * LSTM_K + hcol0 + hh] = (_Float16)h_new;
            y[((size_t)t * LSTM_B + b2) * LSTM_K + hcol0 + hh] = h_new;
        }
        grid_barrier(barrier_ctr, &gen, 32);
    }

    __syncthreads();
    for (int idx = tid; idx < LSTM_B * 16; idx += 256) {
        int b2 = idx >> 4, hh = idx & 15;
        c_state[b2 * LSTM_K + hcol0 + hh] = cbuf[idx];
    }
}

// ---------------------------------------------------------------- final (h,c)
__global__ __launch_bounds__(256) void copy_state_kernel(const float* __restrict__ y_last,
                                                         const float* __restrict__ c_state,
                                                         float* __restrict__ hf,
                                                         float* __restrict__ cf) {
    int i = blockIdx.x * blockDim.x + threadIdx.x;
    if (i < LSTM_B * LSTM_K) {
        hf[i] = y_last[i];
        cf[i] = c_state[i];
    }
}

extern "C" void kernel_launch(void* const* d_in, const int* in_sizes, int n_in,
                              void* d_out, int out_size, void* d_ws, size_t ws_size,
                              hipStream_t stream) {
    (void)in_sizes; (void)n_in; (void)out_size;
    const float* x    = (const float*)d_in[0];   // [2048,32,512]
    const float* w_ih = (const float*)d_in[1];   // [2,2048,512]
    const float* w_hh = (const float*)d_in[2];   // [2,2048,512]
    const float* bias = (const float*)d_in[3];   // [2,2048]
    float* out = (float*)d_out;

    // workspace carve (256B aligned)
    char* ws = (char*)d_ws;
    size_t off = 0;
    auto carve = [&](size_t bytes) -> void* {
        void* p = (void*)(ws + off);
        off += (bytes + 255) & ~(size_t)255;
        return p;
    };
    const size_t wbytes = (size_t)LSTM_L * LSTM_G * LSTM_K * sizeof(_Float16); // 4 MB each
    _Float16* wih16   = (_Float16*)carve(wbytes);
    _Float16* whh16   = (_Float16*)carve(wbytes);
    float*    c_state = (float*)carve((size_t)LSTM_B * LSTM_K * sizeof(float));
    _Float16* h_state = (_Float16*)carve((size_t)LSTM_B * LSTM_K * sizeof(_Float16));
    unsigned* bar     = (unsigned*)carve(256);

    // chunk length: largest that fits remaining ws (deterministic in ws_size)
    int CH = 8;
    {
        size_t remain = ws_size > off ? ws_size - off : 0;
        const int cands[5] = {256, 128, 64, 32, 16};
        for (int i = 0; i < 5; ++i) {
            if ((size_t)cands[i] * LSTM_B * LSTM_G * sizeof(float) <= remain) { CH = cands[i]; break; }
        }
    }
    float* gi = (float*)(ws + off);

    const int nW = LSTM_L * LSTM_G * LSTM_K;
    cvt_f16_kernel<<<(nW + 255) / 256, 256, 0, stream>>>(w_ih, wih16, nW);
    cvt_f16_kernel<<<(nW + 255) / 256, 256, 0, stream>>>(w_hh, whh16, nW);

    float* yout = out;                                    // [2048,32,512]
    float* hf   = out + (size_t)LSTM_T * LSTM_B * LSTM_K; // [2,32,512]
    float* cf   = hf + (size_t)LSTM_L * LSTM_B * LSTM_K;  // [2,32,512]
    const int nch = LSTM_T / CH;

    for (int l = 0; l < LSTM_L; ++l) {
        const _Float16* Wg = wih16 + (size_t)l * LSTM_G * LSTM_K;
        const _Float16* Wr = whh16 + (size_t)l * LSTM_G * LSTM_K;
        const float*    bl = bias + (size_t)l * LSTM_G;
        for (int ch = 0; ch < nch; ++ch) {
            const float* Xc = (l == 0) ? (x + (size_t)ch * CH * LSTM_B * LSTM_K)
                                       : (yout + (size_t)ch * CH * LSTM_B * LSTM_K);
            dim3 ggrid(CH * LSTM_B / 64, LSTM_G / 128);
            gemm_gi_kernel<<<ggrid, 256, 0, stream>>>(Xc, Wg, bl, gi, bar);
            lstm_scan_kernel<<<32, 256, 0, stream>>>(gi, Wr, h_state, c_state,
                                                     yout + (size_t)ch * CH * LSTM_B * LSTM_K,
                                                     bar, CH, ch == 0 ? 1 : 0);
        }
        copy_state_kernel<<<(LSTM_B * LSTM_K + 255) / 256, 256, 0, stream>>>(
            yout + (size_t)(LSTM_T - 1) * LSTM_B * LSTM_K, c_state,
            hf + (size_t)l * LSTM_B * LSTM_K, cf + (size_t)l * LSTM_B * LSTM_K);
    }
}
